// CrossLayerAttention_49022756716934
// MI455X (gfx1250) — compile-verified, matrix operand order checked by reading, not run on hardware
//
#include <hip/hip_runtime.h>
#include <hip/hip_bf16.h>
#include <type_traits>

typedef __bf16 bf16_t;
typedef bf16_t v8bf  __attribute__((ext_vector_type(8)));
typedef bf16_t v16bf __attribute__((ext_vector_type(16)));
typedef float  v8f   __attribute__((ext_vector_type(8)));

static constexpr int CQ    = 256;
static constexpr int BATCH = 16;
static constexpr int HQ = 80, WQ = 80, PQ = HQ * WQ;   // 6400 query pixels
static constexpr int HK = 40, WK = 40, PK = HK * WK;   // 1600 key pixels
static constexpr int HEADS = 4, CH = CQ / HEADS;       // 64 ch / head
static constexpr float SCALE = 0.0625f;                // 1/sqrt(256)

// ---------------- weight f32 -> bf16 ----------------
__global__ void cvt_f32_bf16(const float* __restrict__ in, bf16_t* __restrict__ out, int n) {
  int i = blockIdx.x * blockDim.x + threadIdx.x;
  if (i < n) out[i] = (bf16_t)in[i];
}

// ---------------- WMMA GEMM: Y[b,o,p] = sum_c W[o,c]*X[b,c,p] + bias[o] ----------------
// W: bf16 [256,256] row-major. X: [B,256,P] f32 (converted on load) or bf16.
// Workgroup: 256 threads (8 waves), tile M=64 x N=128, K-step 32 (8 steps, pipelined).
template <bool IN_F32, bool OUT_F32>
__global__ __launch_bounds__(256)
void gemm_conv1x1(const void* __restrict__ Xv, const bf16_t* __restrict__ W,
                  const float* __restrict__ bias, void* __restrict__ Yv, int P) {
  constexpr int BN = 128, BM = 64, BK = 32, LDK = 40;  // LDK pad keeps 16B-aligned frags
  __shared__ alignas(16) bf16_t ldsX[BN * LDK];

  const int t    = threadIdx.x;
  const int b    = blockIdx.z;
  const int p0   = blockIdx.x * BN;
  const int m0   = blockIdx.y * BM;
  const int lane = t & 31, wave = t >> 5;
  const int wm   = wave & 1;        // wave M index (0..1) -> 32 rows each
  const int wn   = wave >> 1;       // wave N index (0..3) -> 32 cols each
  const int half = lane >> 4;       // K-half select per ISA 16-bit layout
  const int lrow = lane & 15;       // M row (A) / N col (B, C/D)

  const float*  Xf = (const float*)Xv;
  const bf16_t* Xb = (const bf16_t*)Xv;

  // staging ownership: one pixel column, 16 consecutive channels per thread.
  // Out-of-range pixels are clamped to a safe address: an out-of-range LDS row n
  // only feeds WMMA output column n, which is masked at the store, so the staged
  // value is irrelevant (finite garbage) and no zero-fill / divergence is needed.
  const int lp   = t & (BN - 1);        // 0..127
  const int lc0  = (t >> 7) * 16;       // 0 or 16
  const int pg   = p0 + lp;
  const int pg_s = pg < P ? pg : 0;
  const size_t xbase = ((size_t)b * 256 + (size_t)lc0) * (size_t)P + (size_t)pg_s;

  using stage_t = typename std::conditional<IN_F32, float, bf16_t>::type;
  stage_t fr[16];

  auto loadX = [&](int k0) {
#pragma unroll
    for (int e = 0; e < 16; e++) {
      size_t idx = xbase + (size_t)(k0 + e) * (size_t)P;
      if constexpr (IN_F32) fr[e] = Xf[idx];
      else                  fr[e] = Xb[idx];
    }
  };

  v8f acc[2][2];
#pragma unroll
  for (int i = 0; i < 2; i++)
#pragma unroll
    for (int j = 0; j < 2; j++)
#pragma unroll
      for (int e = 0; e < 8; e++) acc[i][j][e] = 0.0f;

  loadX(0);

  for (int k0 = 0; k0 < 256; k0 += BK) {
    // pack the staged tile to bf16 (f32 path converts; bf16 path is a pure repack)
    v8bf pa, pb;
#pragma unroll
    for (int e = 0; e < 8; e++) { pa[e] = (bf16_t)fr[e]; pb[e] = (bf16_t)fr[8 + e]; }

    __syncthreads();   // previous tile fully consumed
    *(v8bf*)&ldsX[lp * LDK + lc0]     = pa;   // two ds_store_b128
    *(v8bf*)&ldsX[lp * LDK + lc0 + 8] = pb;
    __syncthreads();   // tile published

    // prefetch next tile while WMMAs run
    if (k0 + BK < 256) loadX(k0 + BK);

    // A fragments (weights) from global: lane<16 -> K {0..7,16..23}, lane>=16 -> K {8..15,24..31}
    v16bf afr[2], bfr[2];
#pragma unroll
    for (int i = 0; i < 2; i++) {
      const bf16_t* ar = W + (size_t)(m0 + wm * 32 + i * 16 + lrow) * 256 + k0 + half * 8;
      v8bf lo = *(const v8bf*)(ar);
      v8bf hi = *(const v8bf*)(ar + 16);
      afr[i] = __builtin_shufflevector(lo, hi, 0,1,2,3,4,5,6,7,8,9,10,11,12,13,14,15);
    }
    // B fragments from LDS (two ds_load_b128 each)
#pragma unroll
    for (int j = 0; j < 2; j++) {
      const bf16_t* br = &ldsX[(wn * 32 + j * 16 + lrow) * LDK + half * 8];
      v8bf lo = *(const v8bf*)(br);
      v8bf hi = *(const v8bf*)(br + 16);
      bfr[j] = __builtin_shufflevector(lo, hi, 0,1,2,3,4,5,6,7,8,9,10,11,12,13,14,15);
    }

#pragma unroll
    for (int i = 0; i < 2; i++)
#pragma unroll
      for (int j = 0; j < 2; j++)
        acc[i][j] = __builtin_amdgcn_wmma_f32_16x16x32_bf16(
            false, afr[i], false, bfr[j], (short)0, acc[i][j], false, false);
  }

  float*  Yf = (float*)Yv;
  bf16_t* Yb = (bf16_t*)Yv;
  // C/D layout: VGPR r -> M = r (lanes 0-15) / M = 8+r (lanes 16-31); N = lane&15
#pragma unroll
  for (int i = 0; i < 2; i++)
#pragma unroll
    for (int j = 0; j < 2; j++) {
      int colg = p0 + wn * 32 + j * 16 + lrow;
      if (colg >= P) continue;
#pragma unroll
      for (int r = 0; r < 8; r++) {
        int rowg = m0 + wm * 32 + i * 16 + half * 8 + r;
        float val = acc[i][j][r] + bias[rowg];
        size_t yi = ((size_t)b * 256 + (size_t)rowg) * (size_t)P + (size_t)colg;
        if constexpr (OUT_F32) Yf[yi] = val;
        else                   Yb[yi] = (bf16_t)val;
      }
    }
}

// ---------------- bilinear coord helper (align_corners=False, scale 2) ----------------
__device__ __forceinline__ void lerp_coord(int o, int lim, int& i0, int& i1, float& f) {
  float s  = o * 0.5f - 0.25f;
  float fl = floorf(s);
  f = s - fl;
  int ii = (int)fl;
  i0 = ii < 0 ? 0 : ii;
  int j = ii + 1;
  i1 = j > lim ? lim : j;
}

// ---------------- score[b,h,x,y] = SCALE * sum_c q * bilinear(k) ----------------
__global__ __launch_bounds__(256)
void score_kernel(const bf16_t* __restrict__ q, const bf16_t* __restrict__ k,
                  float* __restrict__ score) {
  const int x = blockIdx.x;   // 0..79
  const int b = blockIdx.y;   // 0..15
  int ix0, ix1; float fx;
  lerp_coord(x, HK - 1, ix0, ix1, fx);
  for (int task = threadIdx.x; task < HEADS * WQ; task += blockDim.x) {
    int h = task / WQ, y = task - h * WQ;
    int iy0, iy1; float fy;
    lerp_coord(y, WK - 1, iy0, iy1, fy);
    float w00 = (1.0f - fx) * (1.0f - fy), w01 = (1.0f - fx) * fy;
    float w10 = fx * (1.0f - fy),          w11 = fx * fy;
    const bf16_t* qp = q + (((size_t)b * CQ + h * CH) * HQ + x) * WQ + y;
    const bf16_t* kp = k + ((size_t)b * CQ + h * CH) * PK;
    float acc = 0.0f;
#pragma unroll 4
    for (int c = 0; c < CH; c++) {
      const bf16_t* kc = kp + (size_t)c * PK;
      float kv = w00 * (float)kc[ix0 * WK + iy0] + w01 * (float)kc[ix0 * WK + iy1]
               + w10 * (float)kc[ix1 * WK + iy0] + w11 * (float)kc[ix1 * WK + iy1];
      acc += (float)qp[(size_t)c * PQ] * kv;
    }
    score[(((size_t)b * HEADS + h) * HQ + x) * WQ + y] = acc * SCALE;
  }
}

// ---------------- softmax over x per (b,h,y); folds the R2=4 region sum ----------------
__global__ __launch_bounds__(128)
void softmax_kernel(const float* __restrict__ score, float* __restrict__ a4) {
  int y = threadIdx.x;
  if (y >= WQ) return;
  const float* s = score + (size_t)blockIdx.x * PQ + y;
  float m = -3.4e38f;
  for (int x = 0; x < HQ; x++) m = fmaxf(m, s[x * WQ]);
  float sum = 0.0f;
  for (int x = 0; x < HQ; x++) sum += __expf(s[x * WQ] - m);
  float inv = 4.0f / sum;   // x4 == sum over identical R2 region patches
  float* o = a4 + (size_t)blockIdx.x * PQ + y;
  for (int x = 0; x < HQ; x++) o[x * WQ] = __expf(s[x * WQ] - m) * inv;
}

// ---------------- outmat[b,c,x,y] = a4[b,head(c),x,y] * bilinear(v) (bf16) ----------------
__global__ __launch_bounds__(256)
void outmat_kernel(const float* __restrict__ a4, const bf16_t* __restrict__ v,
                   bf16_t* __restrict__ om) {
  size_t idx = (size_t)blockIdx.x * 256 + threadIdx.x;
  if (idx >= (size_t)BATCH * CQ * PQ) return;
  int y = (int)(idx % WQ);
  size_t r = idx / WQ;
  int x = (int)(r % HQ); r /= HQ;
  int c = (int)(r % CQ);
  int b = (int)(r / CQ);
  int ix0, ix1, iy0, iy1; float fx, fy;
  lerp_coord(x, HK - 1, ix0, ix1, fx);
  lerp_coord(y, WK - 1, iy0, iy1, fy);
  const bf16_t* vp = v + ((size_t)b * CQ + c) * PK;
  float vi = (1.0f - fx) * (1.0f - fy) * (float)vp[ix0 * WK + iy0]
           + (1.0f - fx) * fy          * (float)vp[ix0 * WK + iy1]
           + fx * (1.0f - fy)          * (float)vp[ix1 * WK + iy0]
           + fx * fy                   * (float)vp[ix1 * WK + iy1];
  float a = a4[(((size_t)b * HEADS + (c >> 6)) * HQ + x) * WQ + y];
  om[idx] = (bf16_t)(a * vi);
}

// ---------------- launch ----------------
extern "C" void kernel_launch(void* const* d_in, const int* in_sizes, int n_in,
                              void* d_out, int out_size, void* d_ws, size_t ws_size,
                              hipStream_t stream) {
  const float* query = (const float*)d_in[0];
  const float* key   = (const float*)d_in[1];
  const float* wq = (const float*)d_in[2];
  const float* bq = (const float*)d_in[3];
  const float* wk = (const float*)d_in[4];
  const float* bk = (const float*)d_in[5];
  const float* wv = (const float*)d_in[6];
  const float* bv = (const float*)d_in[7];
  const float* wo = (const float*)d_in[8];
  const float* bo = (const float*)d_in[9];

  char* ws = (char*)d_ws;
  size_t off = 0;
  auto alloc = [&](size_t bytes) -> void* {
    void* p = ws + off;
    off += (bytes + 255) & ~(size_t)255;
    return p;
  };
  bf16_t* wqb = (bf16_t*)alloc((size_t)CQ * CQ * 2);
  bf16_t* wkb = (bf16_t*)alloc((size_t)CQ * CQ * 2);
  bf16_t* wvb = (bf16_t*)alloc((size_t)CQ * CQ * 2);
  bf16_t* wob = (bf16_t*)alloc((size_t)CQ * CQ * 2);
  bf16_t* qb  = (bf16_t*)alloc((size_t)BATCH * CQ * PQ * 2);
  bf16_t* kb  = (bf16_t*)alloc((size_t)BATCH * CQ * PK * 2);
  bf16_t* vb  = (bf16_t*)alloc((size_t)BATCH * CQ * PK * 2);
  float*  sc  = (float*) alloc((size_t)BATCH * HEADS * PQ * 4);
  float*  a4  = (float*) alloc((size_t)BATCH * HEADS * PQ * 4);
  bf16_t* om  = (bf16_t*)alloc((size_t)BATCH * CQ * PQ * 2);

  cvt_f32_bf16<<<256, 256, 0, stream>>>(wq, wqb, CQ * CQ);
  cvt_f32_bf16<<<256, 256, 0, stream>>>(wk, wkb, CQ * CQ);
  cvt_f32_bf16<<<256, 256, 0, stream>>>(wv, wvb, CQ * CQ);
  cvt_f32_bf16<<<256, 256, 0, stream>>>(wo, wob, CQ * CQ);

  dim3 blk(256);
  // q/k/v projections (f32 in -> bf16 out)
  gemm_conv1x1<true, false><<<dim3(PQ / 128, 4, BATCH), blk, 0, stream>>>(query, wqb, bq, qb, PQ);
  gemm_conv1x1<true, false><<<dim3((PK + 127) / 128, 4, BATCH), blk, 0, stream>>>(key, wkb, bk, kb, PK);
  gemm_conv1x1<true, false><<<dim3((PK + 127) / 128, 4, BATCH), blk, 0, stream>>>(key, wvb, bv, vb, PK);

  score_kernel  <<<dim3(HQ, BATCH), 256, 0, stream>>>(qb, kb, sc);
  softmax_kernel<<<dim3(BATCH * HEADS), 128, 0, stream>>>(sc, a4);
  outmat_kernel <<<dim3((unsigned)(((size_t)BATCH * CQ * PQ) / 256)), 256, 0, stream>>>(a4, vb, om);

  // output projection (bf16 in -> f32 out, straight to d_out)
  gemm_conv1x1<false, true><<<dim3(PQ / 128, 4, BATCH), blk, 0, stream>>>(om, wob, bo, d_out, PQ);
}
